// FraudGNN_48223892799810
// MI455X (gfx1250) — matrix operand order repaired and checked by reading, not compiled, not run on hardware
//
#include <hip/hip_runtime.h>
#include <hip/hip_bf16.h>
#include <math.h>

typedef __attribute__((ext_vector_type(16))) _Float16 v16h;
typedef __attribute__((ext_vector_type(8)))  float    v8f;

// ---------------------------------------------------------------- utilities

__global__ void fill_zero_kernel(float* __restrict__ p, long long n) {
    long long i = (long long)blockIdx.x * blockDim.x + threadIdx.x;
    if (i < n) p[i] = 0.0f;
}

__global__ void deg_count_kernel(const int* __restrict__ dst, float* __restrict__ deg, int E) {
    int e = blockIdx.x * blockDim.x + threadIdx.x;
    if (e < E) atomicAdd(&deg[dst[e]], 1.0f);
}

// deg -> 1 / max(deg, 1) in place
__global__ void rdeg_kernel(float* __restrict__ deg, int N) {
    int i = blockIdx.x * blockDim.x + threadIdx.x;
    if (i < N) deg[i] = 1.0f / fmaxf(deg[i], 1.0f);
}

// agg[dst[e], :] += x[src[e], :]   (one thread = one edge x 4-channel chunk)
__global__ void scatter_add_kernel(const float* __restrict__ x,
                                   const int* __restrict__ src,
                                   const int* __restrict__ dst,
                                   float* __restrict__ agg, int E, int C) {
    long long idx = (long long)blockIdx.x * blockDim.x + threadIdx.x;
    int chunks = C >> 2;
    long long total = (long long)E * chunks;
    if (idx >= total) return;
    int e  = (int)(idx / chunks);
    int cb = (int)(idx % chunks) << 2;
    int s = src[e], d = dst[e];
    const float4 v = *(const float4*)(x + (size_t)s * C + cb);
    float* ap = agg + (size_t)d * C + cb;
    atomicAdd(ap + 0, v.x);
    atomicAdd(ap + 1, v.y);
    atomicAdd(ap + 2, v.z);
    atomicAdd(ap + 3, v.w);
}

// ---------------------------------------------------------------- fused SAGE layer GEMM
//
// out[M,Ncol] = act( (scale0 ⊙ A0)[M,K1] @ W0[K1,Ncol] + A1[M,K1] @ W1[K1,Ncol] + bias )
//
// One wave computes one 16x16 output tile with v_wmma_f32_16x16x32_f16,
// K stepped by 32, f32 inputs rounded to f16 on load, f32 accumulation.
//
// Per the CDNA5 wave32 layouts (CDNA5 ISA):
//   A 16x32 f16 : lanes 0-15 row M=l, halves {K 0..7, K 16..23};
//                 lanes 16-31 row M=l, halves {K 8..15, K 24..31}
//   B 32x16 f16 : lanes 0-15 col N=l, K {0..7, 16..23}; lanes 16-31 col N=l, K {8..15, 24..31}
//   C/D 16x16 f32: VGPR r -> M=r (lanes 0-15) / M=r+8 (lanes 16-31), N=lane&15
__global__ void sage_gemm_wmma_kernel(const float* __restrict__ A0,
                                      const float* __restrict__ scale0,
                                      const float* __restrict__ A1,
                                      const float* __restrict__ W0,
                                      const float* __restrict__ W1,
                                      const float* __restrict__ bias,
                                      float* __restrict__ out,
                                      int M, int K1, int Ncol, int do_relu) {
    const int lane = threadIdx.x & 31;
    const int wave = threadIdx.x >> 5;
    const int ntN  = Ncol >> 4;
    const int tile = blockIdx.x * (blockDim.x >> 5) + wave;
    const int mt = tile / ntN;
    const int nt = tile % ntN;
    const int m0 = mt << 4;
    if (m0 >= M) return;
    const int n0   = nt << 4;
    const int half = lane >> 4;     // 0: lanes 0-15, 1: lanes 16-31
    const int l15  = lane & 15;
    const int row  = m0 + l15;
    const bool rowOK = row < M;
    const int col  = n0 + l15;

    v8f acc = {};
    #pragma unroll
    for (int s = 0; s < 2; ++s) {
        const float* __restrict__ A = s ? A1 : A0;
        const float* __restrict__ W = s ? W1 : W0;
        const float sc = (s == 0 && rowOK) ? scale0[row] : 1.0f;
        for (int kb = 0; kb < K1; kb += 32) {
            v16h a, b;
            const int ka = kb + half * 8;
            if (rowOK) {
                const float* ap = A + (size_t)row * K1 + ka;
                #pragma unroll
                for (int i = 0; i < 8; ++i) {
                    a[i]     = (_Float16)(ap[i]      * sc);
                    a[i + 8] = (_Float16)(ap[i + 16] * sc);
                }
            } else {
                #pragma unroll
                for (int i = 0; i < 16; ++i) a[i] = (_Float16)0.0f;
            }
            const int kB = kb + half * 8;
            #pragma unroll
            for (int r = 0; r < 16; ++r)
                b[r] = (_Float16)W[(size_t)(kB + r + (r >= 8 ? 8 : 0)) * Ncol + col];
            acc = __builtin_amdgcn_wmma_f32_16x16x32_f16(
                /*neg_a=*/false, a, /*neg_b=*/false, b,
                /*c_mod=*/(short)0, acc, /*reuse_a=*/false, /*reuse_b=*/false);
        }
    }

    const float bv = bias[col];
    #pragma unroll
    for (int r = 0; r < 8; ++r) {
        const int m = m0 + r + half * 8;
        if (m < M) {
            float v = acc[r] + bv;
            if (do_relu) v = fmaxf(v, 0.0f);
            out[(size_t)m * Ncol + col] = v;
        }
    }
}

// ---------------------------------------------------------------- layer 3 + log_softmax
// out[i,:2] = log_softmax( (rdeg[i]*agg3[i]) @ W3l + h2[i] @ W3r + b3 )
__global__ void final_layer_kernel(const float* __restrict__ agg3,
                                   const float* __restrict__ rdeg,
                                   const float* __restrict__ h2,
                                   const float* __restrict__ W3l,
                                   const float* __restrict__ W3r,
                                   const float* __restrict__ b3,
                                   float* __restrict__ out, int N, int K) {
    int i = blockIdx.x * blockDim.x + threadIdx.x;
    if (i >= N) return;
    const float rd = rdeg[i];
    const float* ar = agg3 + (size_t)i * K;
    const float* hr = h2   + (size_t)i * K;
    float o0 = b3[0], o1 = b3[1];
    for (int k = 0; k < K; ++k) {
        const float mv = ar[k] * rd;
        const float hv = hr[k];
        o0 += mv * W3l[2 * k + 0] + hv * W3r[2 * k + 0];
        o1 += mv * W3l[2 * k + 1] + hv * W3r[2 * k + 1];
    }
    const float mx  = fmaxf(o0, o1);
    const float lse = mx + logf(expf(o0 - mx) + expf(o1 - mx));
    out[2 * i + 0] = o0 - lse;
    out[2 * i + 1] = o1 - lse;
}

// ---------------------------------------------------------------- host launch

extern "C" void kernel_launch(void* const* d_in, const int* in_sizes, int n_in,
                              void* d_out, int out_size, void* d_ws, size_t ws_size,
                              hipStream_t stream) {
    (void)n_in; (void)out_size; (void)ws_size;
    const int IN_C = 64, HID = 128;
    const int N = in_sizes[0] / IN_C;   // 50000
    const int E = in_sizes[1] / 2;      // 800000

    const float* x   = (const float*)d_in[0];
    const int*   ei  = (const int*)d_in[1];     // [2, E] int32 (JAX x64 off)
    const int*   src = ei;
    const int*   dst = ei + E;
    const float* W1l = (const float*)d_in[2];
    const float* b1  = (const float*)d_in[3];
    const float* W1r = (const float*)d_in[4];
    const float* W2l = (const float*)d_in[5];
    const float* b2  = (const float*)d_in[6];
    const float* W2r = (const float*)d_in[7];
    const float* W3l = (const float*)d_in[8];
    const float* b3  = (const float*)d_in[9];
    const float* W3r = (const float*)d_in[10];
    float* out = (float*)d_out;

    // workspace: deg/rdeg [N], agg [N,128], h1 [N,128], h2 [N,128]  (~77 MB)
    float* deg = (float*)d_ws;
    float* agg = deg + N;
    float* h1  = agg + (size_t)N * HID;
    float* h2  = h1  + (size_t)N * HID;

    const int T = 256;
    auto blocks = [](long long n, int t) { return (unsigned)((n + t - 1) / t); };

    // degree + reciprocal (shared by all layers)
    fill_zero_kernel<<<blocks(N, T), T, 0, stream>>>(deg, N);
    deg_count_kernel<<<blocks(E, T), T, 0, stream>>>(dst, deg, E);
    rdeg_kernel<<<blocks(N, T), T, 0, stream>>>(deg, N);

    const int WAVES_PER_BLOCK = 4;
    const int GT = WAVES_PER_BLOCK * 32;

    // ---- layer 1: C = 64 -> 128
    fill_zero_kernel<<<blocks((long long)N * IN_C, T), T, 0, stream>>>(agg, (long long)N * IN_C);
    scatter_add_kernel<<<blocks((long long)E * (IN_C / 4), T), T, 0, stream>>>(x, src, dst, agg, E, IN_C);
    {
        long long tiles = (long long)((N + 15) / 16) * (HID / 16);
        sage_gemm_wmma_kernel<<<blocks(tiles, WAVES_PER_BLOCK), GT, 0, stream>>>(
            agg, deg, x, W1l, W1r, b1, h1, N, IN_C, HID, /*relu=*/1);
    }

    // ---- layer 2: C = 128 -> 128
    fill_zero_kernel<<<blocks((long long)N * HID, T), T, 0, stream>>>(agg, (long long)N * HID);
    scatter_add_kernel<<<blocks((long long)E * (HID / 4), T), T, 0, stream>>>(h1, src, dst, agg, E, HID);
    {
        long long tiles = (long long)((N + 15) / 16) * (HID / 16);
        sage_gemm_wmma_kernel<<<blocks(tiles, WAVES_PER_BLOCK), GT, 0, stream>>>(
            agg, deg, h1, W2l, W2r, b2, h2, N, HID, HID, /*relu=*/1);
    }

    // ---- layer 3: C = 128 -> 2, fused with log_softmax
    fill_zero_kernel<<<blocks((long long)N * HID, T), T, 0, stream>>>(agg, (long long)N * HID);
    scatter_add_kernel<<<blocks((long long)E * (HID / 4), T), T, 0, stream>>>(h2, src, dst, agg, E, HID);
    final_layer_kernel<<<blocks(N, T), T, 0, stream>>>(agg, deg, h2, W3l, W3r, b3, out, N, HID);
}
